// Retriever_73572789781210
// MI455X (gfx1250) — compile-verified, matrix operand order checked by reading, not run on hardware
//
#include <hip/hip_runtime.h>

// ---------------------------------------------------------------------------
// Retrieval (IndexFlatL2-style) for MI455X / gfx1250, wave32 + WMMA bf16.
//   x  : [B, K, D] f32     xq : [B, D] f32     xb : [N, D] f32
//   out: [B, K, D] f32 = x + xb[topk_indices]
// Scratch layout in d_ws (floats): [ bsq: N ][ scores: B*N ][ idx: B*K ints ]
// Requires ws_size >= 4*(N + B*N + B*K) bytes (~270 MB).
//
// Roofline: GEMM = 137 GFLOP; xb single-pass = 1.07 GB -> ~46 us at 23.3 TB/s.
// bf16 WMMA (16x16x32) chosen over f32 WMMA (16x16x4): 8x K per instruction;
// ranking tolerates bf16, the gathered docs use exact fp32 xb. Each score
// block covers all 256 queries so xb streams from HBM exactly once.
// ---------------------------------------------------------------------------

typedef __attribute__((ext_vector_type(4)))  float  v4f;
typedef __attribute__((ext_vector_type(8)))  float  v8f;
typedef __attribute__((ext_vector_type(16))) float  v16f;
typedef __attribute__((ext_vector_type(4)))  int    v4i;
typedef __attribute__((ext_vector_type(4)))  __bf16 v4bf;
typedef __attribute__((ext_vector_type(8)))  __bf16 v8bf;
typedef __attribute__((ext_vector_type(16))) __bf16 v16bf;

constexpr int B  = 256;
constexpr int N  = 262144;
constexpr int D  = 1024;
constexpr int K  = 64;

constexpr int NTILE = 128;  // corpus rows per block in score kernel
constexpr int KT    = 32;   // K step per WMMA (bf16 16x16x32)

// ---- gfx1250 async global->LDS path (guarded; fallback = sync staging) ----
#if defined(__has_builtin)
#  if __has_builtin(__builtin_amdgcn_global_load_async_to_lds_b128) && \
      __has_builtin(__builtin_amdgcn_s_wait_asynccnt)
#    define HAVE_ASYNC_LDS 1
#  endif
#endif
#ifndef HAVE_ASYNC_LDS
#  define HAVE_ASYNC_LDS 0
#endif

typedef __attribute__((address_space(1))) v4i* gptr_b128;  // global v4i*
typedef __attribute__((address_space(3))) v4i* lptr_b128;  // LDS v4i*

__device__ __forceinline__ void copy_b128_to_lds(const float* g, float* l) {
#if HAVE_ASYNC_LDS
    // Async DMA: 16B per lane, no VGPR data round-trip, tracked by ASYNCcnt.
    // Integer round-trip avoids illegal generic->AS pointer casts; for LDS
    // the low 32 bits of the generic VA are the LDS offset (ISA aperture).
    __builtin_amdgcn_global_load_async_to_lds_b128(
        (gptr_b128)(unsigned long long)(uintptr_t)g,
        (lptr_b128)(unsigned)(uintptr_t)l,
        /*offset=*/0, /*cpol=*/0);
#else
    *(v4f*)l = *(const v4f*)g;
#endif
}

__device__ __forceinline__ void wait_async_lds() {
#if HAVE_ASYNC_LDS
    __builtin_amdgcn_s_wait_asynccnt(0);
#endif
}

// ---------------------------------------------------------------------------
// Kernel 1: b_sq[n] = sum_d xb[n,d]^2. One wave (32 lanes) per corpus row.
// ---------------------------------------------------------------------------
__global__ void __launch_bounds__(256)
bsq_kernel(const float* __restrict__ xb, float* __restrict__ bsq) {
    const int wave = threadIdx.x >> 5;
    const int lane = threadIdx.x & 31;
    const int n    = blockIdx.x * 8 + wave;
    const float* row = xb + (size_t)n * D;
    float s = 0.0f;
#pragma unroll
    for (int i = 0; i < D / 32; ++i) {
        float v = row[lane + 32 * i];   // coalesced 128B per wave step
        s += v * v;
    }
#pragma unroll
    for (int off = 16; off > 0; off >>= 1)
        s += __shfl_xor(s, off, 32);
    if (lane == 0) bsq[n] = s;
}

// ---------------------------------------------------------------------------
// Kernel 2: scores[b, n] = b_sq[n] - 2 * dot(xq[b], xb[n])  via WMMA bf16.
// 256 threads (8 waves); block covers all 256 queries x NTILE corpus rows.
// Double-buffered LDS, one barrier per K-step:
//   - A (xq) tile: async global->LDS as fp32, converted to bf16 at frag read
//     (each A element is consumed by exactly one wave).
//   - B (xb) tile: global->VGPR->bf16->LDS; bf16 in LDS halves the dominant
//     LDS read traffic (B frags are re-read by all 8 waves). Stores deferred
//     until after the WMMAs so load latency hides under the matrix pipe.
// B fragments are register-pipelined one N-tile ahead so each ds_load pair
// has a WMMA-pair of latency slack; live set stays under 256 VGPRs.
// ---------------------------------------------------------------------------

// 16-bit A/B fragment (16x32) per CDNA5 layout: lane half h holds K runs
// {8h..8h+7} and {16+8h..16+8h+7} of its row.
__device__ __forceinline__ v16bf load_frag_bf(const __bf16* base, int row, int half) {
    const __bf16* p = base + row * KT + 8 * half;
    v8bf lo = *(const v8bf*)(p);
    v8bf hi = *(const v8bf*)(p + 16);
    return __builtin_shufflevector(lo, hi, 0, 1, 2, 3, 4, 5, 6, 7,
                                           8, 9, 10, 11, 12, 13, 14, 15);
}

__device__ __forceinline__ v16bf load_frag_f32(const float* base, int row, int half) {
    const float* p = base + row * KT + 8 * half;
    v8f lo = *(const v8f*)(p);
    v8f hi = *(const v8f*)(p + 16);
    v16f f = __builtin_shufflevector(lo, hi, 0, 1, 2, 3, 4, 5, 6, 7,
                                             8, 9, 10, 11, 12, 13, 14, 15);
    return __builtin_convertvector(f, v16bf);
}

__global__ void __launch_bounds__(256, 1)
score_kernel(const float* __restrict__ xq, const float* __restrict__ xb,
             const float* __restrict__ bsq, float* __restrict__ scores) {
    __shared__ __attribute__((aligned(32))) float  As[2][B * KT];      // 64 KB (fp32)
    __shared__ __attribute__((aligned(16))) __bf16 Bs[2][NTILE * KT];  // 16 KB (bf16)

    const int tid  = threadIdx.x;
    const int wave = tid >> 5;
    const int lane = tid & 31;
    const int half = lane >> 4;
    const int lrow = lane & 15;
    const int n0   = blockIdx.x * NTILE;

    auto stageA = [&](int kb, int buf) {           // 2048 x 16B, 8 per thread
#pragma unroll
        for (int j = 0; j < 8; ++j) {
            int id = tid + 256 * j, row = id >> 3, c = id & 7;
            copy_b128_to_lds(xq + (size_t)row * D + kb + c * 4,
                             &As[buf][row * KT + c * 4]);
        }
    };

    v8f acc[2][8];
#pragma unroll
    for (int mt = 0; mt < 2; ++mt)
#pragma unroll
        for (int nt = 0; nt < 8; ++nt)
#pragma unroll
            for (int r = 0; r < 8; ++r)
                acc[mt][nt][r] = 0.0f;

    // ---- prologue: stage tile 0 into buffer 0 ----
    stageA(0, 0);
#pragma unroll
    for (int j = 0; j < 4; ++j) {                  // B tile: 1024 x 16B
        int id = tid + 256 * j, row = id >> 3, c = id & 7;
        v4f f = *(const v4f*)(xb + (size_t)(n0 + row) * D + c * 4);
        *(v4bf*)(&Bs[0][row * KT + c * 4]) = __builtin_convertvector(f, v4bf);
    }
    wait_async_lds();
    __syncthreads();

    int cur = 0;
    for (int kb = 0; kb < D; kb += KT) {
        const int  nxt  = cur ^ 1;
        const bool more = (kb + KT) < D;

        v4f breg[4];
        if (more) {
            stageA(kb + KT, nxt);                  // async, lands in As[nxt]
#pragma unroll
            for (int j = 0; j < 4; ++j) {          // B globals -> regs (in flight)
                int id = tid + 256 * j, row = id >> 3, c = id & 7;
                breg[j] = *(const v4f*)(xb + (size_t)(n0 + row) * D + (kb + KT) + c * 4);
            }
        }

        // ---- compute tile kb from buffers[cur] ----
        const __bf16* Bc = Bs[cur];
        const float*  Ac = As[cur];

        v16bf afrag[2];
#pragma unroll
        for (int mt = 0; mt < 2; ++mt)
            afrag[mt] = load_frag_f32(Ac, wave * 32 + mt * 16 + lrow, half);

        // B fragments pipelined one N-tile ahead (two register sets ping-pong).
        v16bf bf_cur = load_frag_bf(Bc, lrow, half);
#pragma unroll
        for (int nt = 0; nt < 8; ++nt) {
            v16bf bf_next = bf_cur;
            if (nt < 7)
                bf_next = load_frag_bf(Bc, (nt + 1) * 16 + lrow, half);
#pragma unroll
            for (int mt = 0; mt < 2; ++mt) {
                acc[mt][nt] = __builtin_amdgcn_wmma_f32_16x16x32_bf16(
                    /*neg_a=*/false, afrag[mt], /*neg_b=*/false, bf_cur,
                    /*c_mod=*/(short)0, acc[mt][nt],
                    /*reuse_a=*/false, /*reuse_b=*/false);
            }
            bf_cur = bf_next;
        }

        if (more) {
#pragma unroll
            for (int j = 0; j < 4; ++j) {          // deferred B stores -> Bs[nxt]
                int id = tid + 256 * j, row = id >> 3, c = id & 7;
                *(v4bf*)(&Bs[nxt][row * KT + c * 4]) =
                    __builtin_convertvector(breg[j], v4bf);
            }
            wait_async_lds();                      // own async A copies landed
        }
        __syncthreads();                           // one barrier per K-step
        cur = nxt;
    }

    // D layout (16x16 f32): VGPR r, lanes 0-15 -> M=r, lanes 16-31 -> M=r+8;
    // column = lane % 16. Fold b_sq and the -2 factor at fp32 precision.
#pragma unroll
    for (int nt = 0; nt < 8; ++nt) {
        int n = n0 + nt * 16 + lrow;
        float bq = bsq[n];
#pragma unroll
        for (int mt = 0; mt < 2; ++mt) {
#pragma unroll
            for (int r = 0; r < 8; ++r) {
                int brow = wave * 32 + mt * 16 + r + 8 * half;
                scores[(size_t)brow * N + n] = bq - 2.0f * acc[mt][nt][r];
            }
        }
    }
}

// ---------------------------------------------------------------------------
// Kernel 3: per-query top-K (smallest) selection over N scores.
// One block per query. Each thread streams N/256 scores keeping its own
// top-64 list in LDS (sound: any global top-64 element is in some thread's
// local top-64). Then 64 argmin-extraction rounds, tie-break on lower index
// to match jax.lax.top_k ordering (ascending dist2).
// ---------------------------------------------------------------------------
__global__ void __launch_bounds__(256, 1)
topk_kernel(const float* __restrict__ scores, int* __restrict__ out_idx) {
    __shared__ float candS[256 * 64];   // 64 KB
    __shared__ int   candI[256 * 64];   // 64 KB
    __shared__ float redS[256];
    __shared__ int   redI[256];
    __shared__ int   redOwner[256];
    __shared__ int   redSlot[256];

    const int t = threadIdx.x;
    const int b = blockIdx.x;
    const float* srow = scores + (size_t)b * N;

    float* myS = candS + t * 64;
    int*   myI = candI + t * 64;

    const float INF = __builtin_inff();
    float curMax = -INF;
    int   maxSlot = 0;
    int   count = 0;

    for (int n = t; n < N; n += 256) {
        float s = srow[n];
        if (count < 64) {
            myS[count] = s;
            myI[count] = n;
            if (s > curMax) { curMax = s; maxSlot = count; }
            ++count;
        } else if (s < curMax) {   // strict: equal keeps earlier (lower) index
            myS[maxSlot] = s;
            myI[maxSlot] = n;
            curMax = -INF;
            for (int i = 0; i < 64; ++i) {
                float v = myS[i];
                if (v > curMax) { curMax = v; maxSlot = i; }
            }
        }
    }
    __syncthreads();

    for (int r = 0; r < K; ++r) {
        float bs = INF;
        int bi = 0x7fffffff, bslot = 0;
        for (int i = 0; i < 64; ++i) {
            float v = myS[i];
            int   idx = myI[i];
            if (v < bs || (v == bs && idx < bi)) { bs = v; bi = idx; bslot = i; }
        }
        redS[t] = bs; redI[t] = bi; redOwner[t] = t; redSlot[t] = bslot;
        __syncthreads();
        for (int off = 128; off > 0; off >>= 1) {
            if (t < off) {
                float v2 = redS[t + off];
                int   i2 = redI[t + off];
                if (v2 < redS[t] || (v2 == redS[t] && i2 < redI[t])) {
                    redS[t] = v2; redI[t] = i2;
                    redOwner[t] = redOwner[t + off];
                    redSlot[t]  = redSlot[t + off];
                }
            }
            __syncthreads();
        }
        if (t == 0) out_idx[b * K + r] = redI[0];
        int owner = redOwner[0];
        int slot  = redSlot[0];
        if (t == owner) myS[slot] = INF;   // remove winner from owner's list
        __syncthreads();
    }
}

// ---------------------------------------------------------------------------
// Kernel 4: out[b,k,:] = x[b,k,:] + xb[I[b,k],:]. One block per (b,k) row,
// float4 per thread (256 * 4 = 1024 = D).
// ---------------------------------------------------------------------------
__global__ void __launch_bounds__(256)
gather_add_kernel(const float* __restrict__ x, const float* __restrict__ xb,
                  const int* __restrict__ idx, float* __restrict__ out) {
    const int bk  = blockIdx.x;
    const int doc = idx[bk];
    const int d   = threadIdx.x * 4;
    v4f a = *(const v4f*)(x  + (size_t)bk  * D + d);
    v4f c = *(const v4f*)(xb + (size_t)doc * D + d);
    *(v4f*)(out + (size_t)bk * D + d) = a + c;
}

// ---------------------------------------------------------------------------
extern "C" void kernel_launch(void* const* d_in, const int* in_sizes, int n_in,
                              void* d_out, int out_size, void* d_ws, size_t ws_size,
                              hipStream_t stream) {
    const float* x  = (const float*)d_in[0];   // [B, K, D]
    const float* xq = (const float*)d_in[1];   // [B, D]
    const float* xb = (const float*)d_in[2];   // [N, D]
    (void)in_sizes; (void)n_in; (void)out_size; (void)ws_size;

    float* ws     = (float*)d_ws;
    float* bsq    = ws;                              // N floats
    float* scores = ws + N;                          // B*N floats
    int*   idx    = (int*)(ws + N + (size_t)B * N);  // B*K ints
    float* out    = (float*)d_out;

    bsq_kernel<<<dim3(N / 8), dim3(256), 0, stream>>>(xb, bsq);
    score_kernel<<<dim3(N / NTILE), dim3(256), 0, stream>>>(xq, xb, bsq, scores);
    topk_kernel<<<dim3(B), dim3(256), 0, stream>>>(scores, idx);
    gather_add_kernel<<<dim3(B * K), dim3(256), 0, stream>>>(x, xb, idx, out);
}
